// AttentionHead_16595753631876
// MI455X (gfx1250) — matrix operand order repaired and checked by reading, not compiled, not run on hardware
//
#include <hip/hip_runtime.h>
#include <hip/hip_bf16.h>

// ---------------------------------------------------------------------------
// Types for CDNA5 WMMA
// ---------------------------------------------------------------------------
typedef __bf16 bf_t;
typedef bf_t  bf16x16 __attribute__((ext_vector_type(16)));
typedef float f32x8   __attribute__((ext_vector_type(8)));

#define B_DIM 4
#define T_DIM 2048
#define E_DIM 2048
#define D_DIM 64

// Loads may not sink past this; pins staged-load issue points.
#define SCHED_FENCE() asm volatile("" ::: "memory")

union FragB {
    uint4   q[2];
    bf_t    h[16];
    bf16x16 v;
};

__device__ inline f32x8 zero8() {
    f32x8 z;
#pragma unroll
    for (int i = 0; i < 8; ++i) z[i] = 0.0f;
    return z;
}

__device__ inline f32x8 wmma_bf16(bf16x16 a, bf16x16 b, f32x8 c) {
    // D = A(16x32) * B(32x16) + C  -> v_wmma_f32_16x16x32_bf16
    return __builtin_amdgcn_wmma_f32_16x16x32_bf16(false, a, false, b,
                                                   (short)0, c, false, false);
}

// A operand (16x32, M = lane&15): 8-element runs at k0+8*hl and k0+16+8*hl
__device__ inline bf16x16 load_a_bf16(const bf_t* base, int ld,
                                      int row0, int k0, int lane) {
    int nl = lane & 15, hl = lane >> 4;
    const bf_t* p = base + (size_t)(row0 + nl) * ld + k0 + 8 * hl;
    FragB f;
    f.q[0] = *(const uint4*)(p);
    f.q[1] = *(const uint4*)(p + 16);
    return f.v;
}

// B operand (32x16, N = lane&15): 8-element runs at k0+8*hl and k0+16+8*hl
__device__ inline bf16x16 load_b_bf16(const bf_t* base, int ld,
                                      int n0, int k0, int lane) {
    int nl = lane & 15, hl = lane >> 4;
    const bf_t* p = base + (size_t)(n0 + nl) * ld + k0 + 8 * hl;
    FragB f;
    f.q[0] = *(const uint4*)(p);
    f.q[1] = *(const uint4*)(p + 16);
    return f.v;
}

// Raw fp32 loads for an A fragment (issued early for latency hiding)
__device__ inline void load_a_f32_raw(const float* base, int ld,
                                      int row0, int k0, int lane, float4 r[4]) {
    int nl = lane & 15, hl = lane >> 4;
    const float* p = base + (size_t)(row0 + nl) * ld + k0 + 8 * hl;
    r[0] = *(const float4*)(p);
    r[1] = *(const float4*)(p + 4);
    r[2] = *(const float4*)(p + 16);
    r[3] = *(const float4*)(p + 20);
}

// Convert staged fp32 quads to a bf16 A fragment (v_cvt_pk_bf16_f32)
__device__ inline bf16x16 cvt_a(const float4 r[4]) {
    FragB f;
#pragma unroll
    for (int i = 0; i < 4; ++i) {
        f.h[4 * i + 0] = (bf_t)r[i].x;
        f.h[4 * i + 1] = (bf_t)r[i].y;
        f.h[4 * i + 2] = (bf_t)r[i].z;
        f.h[4 * i + 3] = (bf_t)r[i].w;
    }
    return f.v;
}

// ---------------------------------------------------------------------------
// Kernel 0: fp32 -> bf16 weight conversion
// ---------------------------------------------------------------------------
__global__ void cvt_f32_bf16_kernel(const float* __restrict__ src,
                                    bf_t* __restrict__ dst, int n) {
    int i = blockIdx.x * blockDim.x + threadIdx.x;
    if (i < n) dst[i] = (bf_t)src[i];
}

// ---------------------------------------------------------------------------
// Kernel 1: Q/K/V projections, software-pipelined with scheduling fences.
// One wave computes a 16-row x 64-col tile of all three projections, reusing
// one A (x) fragment per K-step for 12 WMMAs.
// ---------------------------------------------------------------------------
__global__ __launch_bounds__(256)
void proj_qkv_kernel(const float* __restrict__ x,
                     const bf_t* __restrict__ wqb,
                     const bf_t* __restrict__ wkb,
                     const bf_t* __restrict__ wvb,
                     const float* __restrict__ bq,
                     const float* __restrict__ bk,
                     const float* __restrict__ bv,
                     bf_t* __restrict__ qb,
                     bf_t* __restrict__ kb,
                     bf_t* __restrict__ vtb) {
    const int lane = threadIdx.x & 31;
    const int wid  = blockIdx.x * (blockDim.x >> 5) + (threadIdx.x >> 5);
    const int row0 = wid * 16;                         // row in [0, B*T)

    f32x8 accq[4], acck[4], accv[4];
#pragma unroll
    for (int nt = 0; nt < 4; ++nt) { accq[nt] = zero8(); acck[nt] = zero8(); accv[nt] = zero8(); }

    float4 xr[4];
    load_a_f32_raw(x, E_DIM, row0, 0, lane, xr);
    bf16x16 a = cvt_a(xr);

    for (int e0 = 0; e0 < E_DIM; e0 += 32) {
        const int enext = (e0 + 32 < E_DIM) ? e0 + 32 : e0;   // clamped, branchless

        bf16x16 gq[4], gk[4], gv[4];
#pragma unroll
        for (int nt = 0; nt < 4; ++nt) gq[nt] = load_b_bf16(wqb, E_DIM, nt * 16, e0, lane);
#pragma unroll
        for (int nt = 0; nt < 4; ++nt) gk[nt] = load_b_bf16(wkb, E_DIM, nt * 16, e0, lane);
        load_a_f32_raw(x, E_DIM, row0, enext, lane, xr);       // next A, issued now
        __builtin_prefetch(x + (size_t)(row0 + (lane & 15)) * E_DIM + e0 + 64, 0, 3);
        SCHED_FENCE();                         // pin gq/gk/xr loads above Q WMMAs

#pragma unroll
        for (int nt = 0; nt < 4; ++nt) accq[nt] = wmma_bf16(a, gq[nt], accq[nt]);

#pragma unroll
        for (int nt = 0; nt < 4; ++nt) gv[nt] = load_b_bf16(wvb, E_DIM, nt * 16, e0, lane);
        SCHED_FENCE();                         // pin gv loads above K WMMAs

#pragma unroll
        for (int nt = 0; nt < 4; ++nt) acck[nt] = wmma_bf16(a, gk[nt], acck[nt]);

        const bf16x16 anext = cvt_a(xr);       // cvts co-execute with WMMAs

#pragma unroll
        for (int nt = 0; nt < 4; ++nt) accv[nt] = wmma_bf16(a, gv[nt], accv[nt]);
        a = anext;
    }

    const int nl = lane & 15, hl = lane >> 4;
    const int b  = row0 / T_DIM;
    const int tb = (row0 % T_DIM) + 8 * hl;
#pragma unroll
    for (int nt = 0; nt < 4; ++nt) {
        const int col = nt * 16 + nl;
        const float bqv = bq[col], bkv = bk[col], bvv = bv[col];
        uint4 packv;
        bf_t* pv = (bf_t*)&packv;
#pragma unroll
        for (int r = 0; r < 8; ++r) {
            const int grow = row0 + r + 8 * hl;
            qb[(size_t)grow * D_DIM + col] = (bf_t)(accq[nt][r] + bqv);
            kb[(size_t)grow * D_DIM + col] = (bf_t)(acck[nt][r] + bkv);
            pv[r] = (bf_t)(accv[nt][r] + bvv);
        }
        *(uint4*)(vtb + ((size_t)b * D_DIM + col) * T_DIM + tb) = packv;
    }
}

// ---------------------------------------------------------------------------
// Kernel 2: causal flash attention.  One wave per 16-query tile; key loop in
// blocks of 32 with online softmax.  Next block's K fragments load under the
// current P*V WMMAs; V fragments are fenced above the softmax VALU block.
// ---------------------------------------------------------------------------
__global__ __launch_bounds__(32)
void flash_attn_kernel(const bf_t* __restrict__ qb,
                       const bf_t* __restrict__ kb,
                       const bf_t* __restrict__ vtb,
                       float* __restrict__ out) {
    __shared__ __align__(16) bf_t ldsP[16 * 32];

    const int lane = threadIdx.x & 31;
    const int nl = lane & 15, hl = lane >> 4;
    const int qt = blockIdx.x;                 // 0 .. B*T/16-1
    const int b  = qt >> 7;                    // 128 tiles per batch
    const int i0 = (qt & 127) << 4;

    const bf_t* qbb = qb  + (size_t)b * T_DIM * D_DIM;
    const bf_t* kbb = kb  + (size_t)b * T_DIM * D_DIM;
    const bf_t* vbb = vtb + (size_t)b * D_DIM * T_DIM;

    const bf16x16 aq0 = load_a_bf16(qbb, D_DIM, i0, 0,  lane);
    const bf16x16 aq1 = load_a_bf16(qbb, D_DIM, i0, 32, lane);

    f32x8 o[4];
#pragma unroll
    for (int nt = 0; nt < 4; ++nt) o[nt] = zero8();
    float rmax[8], rsum[8];
#pragma unroll
    for (int r = 0; r < 8; ++r) { rmax[r] = -__builtin_inff(); rsum[r] = 0.0f; }

    const float scale = 0.125f;                // 1/sqrt(64)
    const int jend = i0 + 16;

    // preload first K block
    bf16x16 kf[4];
    kf[0] = load_b_bf16(kbb, D_DIM, 0,  0,  lane);
    kf[1] = load_b_bf16(kbb, D_DIM, 0,  32, lane);
    kf[2] = load_b_bf16(kbb, D_DIM, 16, 0,  lane);
    kf[3] = load_b_bf16(kbb, D_DIM, 16, 32, lane);

    for (int j0 = 0; j0 < jend; j0 += 32) {
        // ---- S = (Q K^T), K fragments already in flight ----
        f32x8 s0 = zero8(), s1 = zero8();
        s0 = wmma_bf16(aq0, kf[0], s0);
        s0 = wmma_bf16(aq1, kf[1], s0);
        s1 = wmma_bf16(aq0, kf[2], s1);
        s1 = wmma_bf16(aq1, kf[3], s1);

        // ---- issue V fragments; fence pins them above the softmax VALU ----
        bf16x16 vf[4];
#pragma unroll
        for (int nt = 0; nt < 4; ++nt)
            vf[nt] = load_b_bf16(vbb, T_DIM, nt * 16, j0, lane);
        SCHED_FENCE();

        const bool boundary = (j0 + 31 > i0);  // wave-uniform

        // ---- online softmax over the 32-key block ----
#pragma unroll
        for (int r = 0; r < 8; ++r) {
            float v0 = s0[r] * scale;
            float v1 = s1[r] * scale;
            if (boundary) {
                const int irow = i0 + r + 8 * hl;
                v0 = (j0 + nl      > irow) ? -__builtin_inff() : v0;
                v1 = (j0 + 16 + nl > irow) ? -__builtin_inff() : v1;
            }
            float bm = fmaxf(v0, v1);
#pragma unroll
            for (int off = 1; off < 16; off <<= 1)
                bm = fmaxf(bm, __shfl_xor(bm, off));
            const float nm   = fmaxf(rmax[r], bm);
            const float corr = __expf(rmax[r] - nm);
            const float p0   = __expf(v0 - nm);
            const float p1   = __expf(v1 - nm);
            float ls = p0 + p1;
#pragma unroll
            for (int off = 1; off < 16; off <<= 1)
                ls += __shfl_xor(ls, off);
            rsum[r] = rsum[r] * corr + ls;
            rmax[r] = nm;
            o[0][r] *= corr; o[1][r] *= corr; o[2][r] *= corr; o[3][r] *= corr;
            ldsP[(r + 8 * hl) * 32 + nl]      = (bf_t)p0;
            ldsP[(r + 8 * hl) * 32 + nl + 16] = (bf_t)p1;
        }
        __syncthreads();                        // single-wave WG: just orders LDS

        // ---- reload P as 16x32 A operand ----
        FragB fp;
        {
            const bf_t* p = ldsP + nl * 32 + 8 * hl;
            fp.q[0] = *(const uint4*)(p);
            fp.q[1] = *(const uint4*)(p + 16);
        }
        __syncthreads();

        // ---- next block's K fragments load under the P*V WMMAs ----
        const int jn = (j0 + 32 < jend) ? j0 + 32 : j0;        // clamped
        bf16x16 kfn[4];
        kfn[0] = load_b_bf16(kbb, D_DIM, jn,      0,  lane);
        kfn[1] = load_b_bf16(kbb, D_DIM, jn,      32, lane);
        kfn[2] = load_b_bf16(kbb, D_DIM, jn + 16, 0,  lane);
        kfn[3] = load_b_bf16(kbb, D_DIM, jn + 16, 32, lane);
        SCHED_FENCE();

        // ---- O += P * V  (V^T rows are contiguous in t) ----
#pragma unroll
        for (int nt = 0; nt < 4; ++nt)
            o[nt] = wmma_bf16(fp.v, vf[nt], o[nt]);

#pragma unroll
        for (int nt = 0; nt < 4; ++nt) kf[nt] = kfn[nt];
    }

    // ---- normalize and store fp32 output [B,T,64] ----
    float inv[8];
#pragma unroll
    for (int r = 0; r < 8; ++r) inv[r] = 1.0f / rsum[r];
#pragma unroll
    for (int nt = 0; nt < 4; ++nt)
#pragma unroll
        for (int r = 0; r < 8; ++r)
            out[((size_t)b * T_DIM + i0 + r + 8 * hl) * D_DIM + nt * 16 + nl] =
                o[nt][r] * inv[r];
}

// ---------------------------------------------------------------------------
// Launch
// ---------------------------------------------------------------------------
extern "C" void kernel_launch(void* const* d_in, const int* in_sizes, int n_in,
                              void* d_out, int out_size, void* d_ws, size_t ws_size,
                              hipStream_t stream) {
    const float* x  = (const float*)d_in[0];
    const float* Wq = (const float*)d_in[1];
    const float* bq = (const float*)d_in[2];
    const float* Wk = (const float*)d_in[3];
    const float* bk = (const float*)d_in[4];
    const float* Wv = (const float*)d_in[5];
    const float* bv = (const float*)d_in[6];
    float* out = (float*)d_out;

    const size_t BTD = (size_t)B_DIM * T_DIM * D_DIM;   // 524288
    const size_t WSZ = (size_t)D_DIM * E_DIM;           // 131072
    bf_t* qb  = (bf_t*)d_ws;
    bf_t* kb  = qb  + BTD;
    bf_t* vtb = kb  + BTD;
    bf_t* wqb = vtb + BTD;
    bf_t* wkb = wqb + WSZ;
    bf_t* wvb = wkb + WSZ;

    // 0) convert weights fp32 -> bf16
    {
        dim3 blk(256), grd((WSZ + 255) / 256);
        cvt_f32_bf16_kernel<<<grd, blk, 0, stream>>>(Wq, wqb, (int)WSZ);
        cvt_f32_bf16_kernel<<<grd, blk, 0, stream>>>(Wk, wkb, (int)WSZ);
        cvt_f32_bf16_kernel<<<grd, blk, 0, stream>>>(Wv, wvb, (int)WSZ);
    }

    // 1) projections: 512 row-tiles, 8 waves per block
    proj_qkv_kernel<<<64, 256, 0, stream>>>(x, wqb, wkb, wvb, bq, bk, bv,
                                            qb, kb, vtb);

    // 2) flash attention: one wave per 16-query tile
    flash_attn_kernel<<<B_DIM * T_DIM / 16, 32, 0, stream>>>(qb, kb, vtb, out);
}
